// PW_classifier_14104672600544
// MI455X (gfx1250) — matrix operand order repaired and checked by reading, not compiled, not run on hardware
//
#include <hip/hip_runtime.h>
#include <stdint.h>
#include <stddef.h>

// ---------- types ----------
typedef __attribute__((ext_vector_type(16))) __bf16 bf16x16;
typedef __attribute__((ext_vector_type(8)))  __bf16 bf16x8;
typedef __attribute__((ext_vector_type(8)))  float  f32x8;
typedef __attribute__((ext_vector_type(4)))  unsigned u32x4;
typedef __attribute__((ext_vector_type(8)))  int      i32x8;
typedef __attribute__((ext_vector_type(4)))  int      i32x4;
typedef __attribute__((__vector_size__(16))) int      v4i_vs;

#define DEVI __device__ __forceinline__

// ---------- CDNA5 feature detection (device pass only) ----------
#if defined(__AMDGCN__) && __has_builtin(__builtin_amdgcn_global_load_async_to_lds_b128)
#define HAVE_ASYNC_LDS 1
#else
#define HAVE_ASYNC_LDS 0
#endif
#if defined(__AMDGCN__) && __has_builtin(__builtin_amdgcn_tensor_load_to_lds)
#define HAVE_TDM 1
#else
#define HAVE_TDM 0
#endif
#if defined(__AMDGCN__) && __has_builtin(__builtin_amdgcn_s_wait_asynccnt)
#define HAVE_WAIT_ASYNC 1
#else
#define HAVE_WAIT_ASYNC 0
#endif

DEVI __bf16 f2bf(float f) {
  unsigned u = __float_as_uint(f);
  unsigned r = (u + 0x7FFFu + ((u >> 16) & 1u)) >> 16;
  unsigned short h = (unsigned short)r;
  return __builtin_bit_cast(__bf16, h);
}
DEVI bf16x8 bf8_zero() {
  union { bf16x8 v; unsigned u[4]; } z;
  z.u[0] = z.u[1] = z.u[2] = z.u[3] = 0u;
  return z.v;
}

// 16-byte global -> LDS copy (async on CDNA5, sync fallback otherwise)
#if HAVE_ASYNC_LDS
typedef __attribute__((address_space(1))) v4i_vs gmem_v4i;
typedef __attribute__((address_space(3))) v4i_vs lds_v4i;
#endif
DEVI void async_copy16(const void* g, void* l) {
#if HAVE_ASYNC_LDS
  __builtin_amdgcn_global_load_async_to_lds_b128(
      (gmem_v4i*)(uintptr_t)g, (lds_v4i*)l, 0, 0);
#else
  *(bf16x8*)l = *(const bf16x8*)g;
#endif
}
DEVI void wait_async_all() {
#if HAVE_WAIT_ASYNC
  __builtin_amdgcn_s_wait_asynccnt(0);
#elif HAVE_ASYNC_LDS
  asm volatile("s_wait_asynccnt 0" ::: "memory");
#endif
}
DEVI void wait_tensor_all() {
#if HAVE_TDM
  __builtin_amdgcn_s_wait_tensorcnt(0);
#endif
}

#if HAVE_TDM
DEVI unsigned lds_off(const void* p) {
  return (unsigned)(uintptr_t)(__attribute__((address_space(3))) const void*)p;
}
// TDM: 2D bf16 tile load, tile = 128 rows x 32 halves (64B), row stride = lda halves.
// pad_enable inserts 4 DWORDs (16B) after every 16 DWORDs (64B) -> matches LDS [128][40].
DEVI void tdm_load_tile_bf16(unsigned lds_byte_off, const void* gptr, int lda_halves) {
  unsigned long long ga = (unsigned long long)(uintptr_t)gptr;
  u32x4 g0;
  g0[0] = 1u;                                            // count=1 (valid descriptor)
  g0[1] = lds_byte_off;                                  // lds_addr
  g0[2] = (unsigned)(ga & 0xffffffffull);                // global_addr[31:0]
  g0[3] = (unsigned)((ga >> 32) & 0x01ffffffull) | (2u << 30);  // addr[56:32] | type=2
  i32x8 g1;
  g1[0] = (int)((1u << 16) | (1u << 20) | (3u << 22) | (3u << 25));
         // data_size=2B | pad_enable | pad_interval=16dw | pad_amount=4dw
  g1[1] = (int)0xffff0000u;                              // tensor_dim0 lo16 (huge: interior only)
  g1[2] = (int)(0x7fffu | (0xffffu << 16));              // dim0 hi16 | tensor_dim1 lo16
  g1[3] = (int)(0x7fffu | (32u << 16));                  // dim1 hi16 | tile_dim0 = 32
  g1[4] = (int)(128u);                                   // tile_dim1 = 128, tile_dim2 = 0
  g1[5] = lda_halves;                                    // tensor_dim0_stride lo32
  g1[6] = 0;
  g1[7] = 0;
  i32x4 gz = {0, 0, 0, 0};
#if __clang_major__ >= 23
  i32x8 gz8 = {0, 0, 0, 0, 0, 0, 0, 0};
  __builtin_amdgcn_tensor_load_to_lds(g0, g1, gz, gz, gz8, 0);
#else
  __builtin_amdgcn_tensor_load_to_lds(g0, g1, gz, gz, 0);
#endif
}
#endif // HAVE_TDM

// =====================================================================
// WMMA GEMM: C[M,N] = p(A) @ B^T-layout, bf16 operands, f32 accumulate.
// AMODE 0: A bf16 [M,K] row-major (lda)       -> TDM tile load (interior)
// AMODE 1: A f32  [M,K]; p = exp(a-smax[row])*sinv[row] (fused row softmax)
// AMODE 2: A f32  [K,M] read transposed; stats indexed by output row
// B is ALWAYS bf16 [N,K] row-major (ldb)      -> per-lane async b128 (interior)
// Block 256 thr = 8 waves; block tile 128x128, K-tile 32.
// =====================================================================
template<int AMODE>
__global__ __launch_bounds__(256)
void gemm_wmma(const void* __restrict__ Av, const __bf16* __restrict__ B,
               float* __restrict__ C, int M, int N, int K,
               int lda, int ldb, int ldc,
               const float* __restrict__ bias, int relu,
               const float* __restrict__ smax, const float* __restrict__ sinv)
{
  __shared__ __bf16 As[128][40];
  __shared__ __bf16 Bs[128][40];
  const int tid  = threadIdx.x;
  const int lane = tid & 31;
  const int wave = tid >> 5;
  const int wm = wave >> 2;   // 0..1  (64-row wave tile)
  const int wn = wave & 3;    // 0..3  (32-col wave tile)
  const int m0 = blockIdx.y * 128;
  const int n0 = blockIdx.x * 128;

  const __bf16* Ab = (const __bf16*)Av;
  const float*  Af = (const float*)Av;

  f32x8 acc[4][2];
#pragma unroll
  for (int i = 0; i < 4; ++i)
#pragma unroll
    for (int j = 0; j < 2; ++j)
#pragma unroll
      for (int r = 0; r < 8; ++r) acc[i][j][r] = 0.0f;

  for (int k0 = 0; k0 < K; k0 += 32) {
    const bool intA = (AMODE == 0) && (m0 + 128 <= M) && (k0 + 32 <= K);
    const bool intB = (n0 + 128 <= N) && (k0 + 32 <= K);

    // ---------------- A tile -> As[m][k] ----------------
    if (AMODE == 0) {
      if (intA) {
#if HAVE_TDM
        if (wave == 0)
          tdm_load_tile_bf16(lds_off(&As[0][0]), Ab + (size_t)m0 * lda + k0, lda);
#else
#pragma unroll
        for (int it = 0; it < 2; ++it) {
          int s  = it * 256 + tid;      // 512 slots: 128 rows x 4 chunks of 8 halves
          int r  = s >> 2;
          int cg = (s & 3) * 8;
          async_copy16(Ab + (size_t)(m0 + r) * lda + (k0 + cg), &As[r][cg]);
        }
#endif
      } else {
#pragma unroll
        for (int it = 0; it < 2; ++it) {
          int s  = it * 256 + tid;
          int r  = s >> 2;
          int cg = (s & 3) * 8;
          int gr = m0 + r, gc = k0 + cg;
          bf16x8 val = bf8_zero();
          if (gr < M) {
            if (gc + 7 < K) {
              val = *(const bf16x8*)(Ab + (size_t)gr * lda + gc);
            } else {
#pragma unroll
              for (int i = 0; i < 8; ++i)
                if (gc + i < K) val[i] = Ab[(size_t)gr * lda + gc + i];
            }
          }
          *(bf16x8*)&As[r][cg] = val;
        }
      }
    } else if (AMODE == 1) {
      int r  = tid >> 1;                // 128 rows x 2 chunks of 16 floats
      int cg = (tid & 1) * 16;
      int gr = m0 + r;
      float mrow = (gr < M) ? smax[gr] : 0.f;
      float inv  = (gr < M) ? sinv[gr] : 0.f;
      if (gr < M && k0 + 32 < K)
        __builtin_prefetch(Af + (size_t)gr * lda + k0 + 32 + cg, 0, 1);
      union { bf16x8 v[2]; __bf16 h[16]; } t;
#pragma unroll
      for (int i = 0; i < 16; ++i) {
        int gc = k0 + cg + i;
        float a = (gr < M && gc < K) ? Af[(size_t)gr * lda + gc] : -3.4e38f;
        t.h[i] = f2bf(__expf(a - mrow) * inv);
      }
      *(bf16x8*)&As[r][cg]     = t.v[0];
      *(bf16x8*)&As[r][cg + 8] = t.v[1];
    } else { // AMODE == 2 : A element (mm,kk) = Af[kk][mm], stats by output row
      int jj  = tid >> 3;               // 32 k-rows
      int ii0 = (tid & 7) * 16;         // 8 chunks of 16 along m
      int gj  = k0 + jj;
#pragma unroll
      for (int i = 0; i < 16; ++i) {
        int gi = m0 + ii0 + i;
        float mm  = (gi < M) ? smax[gi] : 0.f;
        float inv = (gi < M) ? sinv[gi] : 0.f;
        float a = (gj < K && gi < M) ? Af[(size_t)gj * lda + gi] : -3.4e38f;
        As[ii0 + i][jj] = f2bf(__expf(a - mm) * inv);
      }
    }

    // ---------------- B tile -> Bs[n][k] (B is [N,K] row-major) ----------------
    if (intB) {
#pragma unroll
      for (int it = 0; it < 2; ++it) {
        int s  = it * 256 + tid;        // 512 slots: 128 rows x 4 chunks of 8 halves
        int r  = s >> 2;
        int cg = (s & 3) * 8;
        async_copy16(B + (size_t)(n0 + r) * ldb + (k0 + cg), &Bs[r][cg]);
      }
    } else {
#pragma unroll
      for (int it = 0; it < 2; ++it) {
        int s  = it * 256 + tid;
        int r  = s >> 2;
        int cg = (s & 3) * 8;
        int gn = n0 + r, gc = k0 + cg;
        bf16x8 val = bf8_zero();
        if (gn < N) {
          if (gc + 7 < K) {
            val = *(const bf16x8*)(B + (size_t)gn * ldb + gc);
          } else {
#pragma unroll
            for (int i = 0; i < 8; ++i)
              if (gc + i < K) val[i] = B[(size_t)gn * ldb + gc + i];
          }
        }
        *(bf16x8*)&Bs[r][cg] = val;
      }
    }

    wait_tensor_all();   // TENSORcnt: wave0's TDM descriptor done
    wait_async_all();    // ASYNCcnt: this wave's async b128 copies done
    __syncthreads();

    // ---------------- fragments + WMMA ----------------
    union U16 { bf16x16 v; bf16x8 h[2]; };
    U16 afr[4], bfr[2];
    const int li = lane & 15;
    const int K0 = (lane < 16) ? 0 : 8;   // A frag: VGPR0-3 = K0..K0+7, VGPR4-7 = K0+16..K0+23
#pragma unroll
    for (int tm = 0; tm < 4; ++tm) {
      int r = wm * 64 + tm * 16 + li;
      afr[tm].h[0] = *(const bf16x8*)&As[r][K0];
      afr[tm].h[1] = *(const bf16x8*)&As[r][K0 + 16];
    }
    const int Kb = (lane < 16) ? 0 : 8;   // B frag: VGPR0-3 = Kb..Kb+7, VGPR4-7 = Kb+16..Kb+23
#pragma unroll
    for (int tn = 0; tn < 2; ++tn) {
      int n = wn * 32 + tn * 16 + li;
      bfr[tn].h[0] = *(const bf16x8*)&Bs[n][Kb];
      bfr[tn].h[1] = *(const bf16x8*)&Bs[n][Kb + 16];
    }
#pragma unroll
    for (int tm = 0; tm < 4; ++tm)
#pragma unroll
      for (int tn = 0; tn < 2; ++tn)
        acc[tm][tn] = __builtin_amdgcn_wmma_f32_16x16x32_bf16(
            false, afr[tm].v, false, bfr[tn].v, (short)0, acc[tm][tn], false, false);
    __syncthreads();
  }

  // ---------------- epilogue ----------------
#pragma unroll
  for (int tm = 0; tm < 4; ++tm) {
#pragma unroll
    for (int tn = 0; tn < 2; ++tn) {
      int col   = n0 + wn * 32 + tn * 16 + (lane & 15);
      int rbase = m0 + wm * 64 + tm * 16 + ((lane < 16) ? 0 : 8);
      if (col < N) {
        float bv = bias ? bias[col] : 0.f;
#pragma unroll
        for (int r = 0; r < 8; ++r) {
          int row = rbase + r;
          if (row < M) {
            float v = acc[tm][tn][r] + bv;
            if (relu) v = fmaxf(v, 0.f);
            C[(size_t)row * ldc + col] = v;
          }
        }
      }
    }
  }
}

// =====================================================================
// Small kernels
// =====================================================================
__global__ void k_f2bf(const float* __restrict__ in, __bf16* __restrict__ out, size_t n) {
  size_t i = (size_t)blockIdx.x * 256 + threadIdx.x;
  if (i < n) out[i] = f2bf(in[i]);
}

// f32 [R,C] -> bf16 [C,R]  (build transposed B operands)
__global__ void k_f2bfT(const float* __restrict__ in, __bf16* __restrict__ out, int R, int C) {
  size_t i = (size_t)blockIdx.x * 256 + threadIdx.x;
  if (i >= (size_t)R * C) return;
  int r = (int)(i / C), c = (int)(i % C);
  out[(size_t)c * R + r] = f2bf(in[i]);
}

__global__ void k_add(const float* a, const float* b, float* o, size_t n) {
  size_t i = (size_t)blockIdx.x * 256 + threadIdx.x;
  if (i < n) o[i] = a[i] + b[i];
}

__global__ void k_concat_bf(const float* __restrict__ a, const float* __restrict__ b,
                            __bf16* __restrict__ out, int rows, int D) {
  size_t i = (size_t)blockIdx.x * 256 + threadIdx.x;
  size_t total = (size_t)rows * 2 * D;
  if (i >= total) return;
  int r = (int)(i / (2 * D));
  int d = (int)(i % (2 * D));
  float v = (d < D) ? a[(size_t)r * D + d] : b[(size_t)r * D + (d - D)];
  out[i] = f2bf(v);
}

// wave-per-row GEMV: out[r] = f(dot(W[r,:], v) + bias0)
__global__ void k_matvec(const float* __restrict__ W, const float* __restrict__ v,
                         float* __restrict__ out, int rows, int cols,
                         const float* __restrict__ bias, int sig) {
  int w = blockIdx.x * 8 + (threadIdx.x >> 5);
  int lane = threadIdx.x & 31;
  if (w >= rows) return;
  float s = 0.f;
  for (int c = lane; c < cols; c += 32) s += W[(size_t)w * cols + c] * v[c];
#pragma unroll
  for (int o = 16; o > 0; o >>= 1) s += __shfl_xor(s, o, 32);
  if (lane == 0) {
    if (bias) s += bias[0];
    if (sig) s = 1.f / (1.f + __expf(-s));
    out[w] = s;
  }
}

__global__ void k_rowstats(const float* __restrict__ A, int cols, int lda,
                           float* __restrict__ rmax, float* __restrict__ rinv) {
  int row = blockIdx.x;
  int tid = threadIdx.x;
  __shared__ float red[256];
  const float* a = A + (size_t)row * lda;
  float m = -3.4e38f;
  for (int c = tid; c < cols; c += 256) m = fmaxf(m, a[c]);
  red[tid] = m; __syncthreads();
  for (int s = 128; s > 0; s >>= 1) { if (tid < s) red[tid] = fmaxf(red[tid], red[tid + s]); __syncthreads(); }
  m = red[0]; __syncthreads();
  float sum = 0.f;
  for (int c = tid; c < cols; c += 256) sum += __expf(a[c] - m);
  red[tid] = sum; __syncthreads();
  for (int s = 128; s > 0; s >>= 1) { if (tid < s) red[tid] += red[tid + s]; __syncthreads(); }
  if (tid == 0) { rmax[row] = m; rinv[row] = 1.f / red[0]; }
}

__global__ void k_colstats(const float* __restrict__ A, int rows, int cols, int lda,
                           float* __restrict__ cmax, float* __restrict__ cinv) {
  int c = blockIdx.x * 256 + threadIdx.x;
  if (c >= cols) return;
  float m = -3.4e38f;
  for (int r = 0; r < rows; ++r) m = fmaxf(m, A[(size_t)r * lda + c]);
  float s = 0.f;
  for (int r = 0; r < rows; ++r) s += __expf(A[(size_t)r * lda + c] - m);
  cmax[c] = m; cinv[c] = 1.f / s;
}

__global__ void k_colmean_relu(const float* __restrict__ h, float* __restrict__ c, int N, int D) {
  int d = blockIdx.x * 256 + threadIdx.x;
  if (d >= D) return;
  float s = 0.f;
  for (int n = 0; n < N; ++n) s += h[(size_t)n * D + d];
  c[d] = fmaxf(s / (float)N, 0.f);
}

__global__ void k_asym(const float* __restrict__ A, float* __restrict__ S, int D) {
  int i = blockIdx.x * 256 + threadIdx.x;
  if (i >= D * D) return;
  int r = i / D, c = i % D;
  S[i] = 0.5f * (A[i] + A[(size_t)c * D + r]);
}

// GCN attention: w = softmax_k(vq[adj]+e·Weq); out = relu((gather + wsum_e·We)/16 + Zc + b)
__global__ void k_gcn_attn(const float* __restrict__ vWn, const float* __restrict__ vq,
                           const float* __restrict__ Zc, const int* __restrict__ adj,
                           const float* __restrict__ e, const float* __restrict__ weq,
                           const float* __restrict__ We, const float* __restrict__ bias,
                           float* __restrict__ out, int N, int D, int permMul) {
  int n = blockIdx.x;
  int r = permMul ? (int)(((long long)n * permMul + 13) % N) : n;
  __shared__ float w[16];
  __shared__ int sadj[16];
  __shared__ float se[16][2];
  int tid = threadIdx.x;
  if (tid < 32) {
    float s = -3.4e38f; int a = 0; float ea = 0.f, eb = 0.f;
    if (tid < 16) {
      a  = adj[(size_t)r * 16 + tid];
      ea = e[((size_t)r * 16 + tid) * 2 + 0];
      eb = e[((size_t)r * 16 + tid) * 2 + 1];
      s  = vq[a] + ea * weq[0] + eb * weq[1];
    }
    float m = s;
#pragma unroll
    for (int o = 8; o > 0; o >>= 1) m = fmaxf(m, __shfl_xor(m, o, 32));
    float ex = (tid < 16) ? __expf(s - m) : 0.f;
    float sum = ex;
#pragma unroll
    for (int o = 8; o > 0; o >>= 1) sum += __shfl_xor(sum, o, 32);
    if (tid < 16) { w[tid] = ex / sum; sadj[tid] = a; se[tid][0] = ea; se[tid][1] = eb; }
  }
  __syncthreads();
  float we0 = 0.f, we1 = 0.f;
#pragma unroll
  for (int k = 0; k < 16; ++k) { we0 += w[k] * se[k][0]; we1 += w[k] * se[k][1]; }
  for (int d = tid; d < D; d += blockDim.x) {
    float acc = 0.f;
#pragma unroll
    for (int k = 0; k < 16; ++k) acc += w[k] * vWn[(size_t)sadj[k] * D + d];
    acc += we0 * We[d] + we1 * We[D + d];
    float z = acc * (1.f / 16.f) + Zc[(size_t)n * D + d] + bias[d];
    out[(size_t)n * D + d] = fmaxf(z, 0.f);
  }
}

// rows 0..P-1: [h0[e0], h1[e1]]; rows P..2P-1: [h1[e1], h0[e0]]  (bf16, width 2D)
__global__ void k_head_gather(const float* __restrict__ h0, const float* __restrict__ h1,
                              const int* __restrict__ ex, __bf16* __restrict__ g, int P, int D) {
  size_t i = (size_t)blockIdx.x * 256 + threadIdx.x;
  size_t total = (size_t)2 * P * 2 * D;
  if (i >= total) return;
  int row = (int)(i / (2 * D));
  int d   = (int)(i % (2 * D));
  int p = row % P;
  bool second = row >= P;
  int e0 = ex[p * 2 + 0], e1 = ex[p * 2 + 1];
  const float* fst = second ? (h1 + (size_t)e1 * D) : (h0 + (size_t)e0 * D);
  const float* snd = second ? (h0 + (size_t)e0 * D) : (h1 + (size_t)e1 * D);
  float v = (d < D) ? fst[d] : snd[d - D];
  g[i] = f2bf(v);
}

__global__ void k_head_final(const float* __restrict__ t, float* __restrict__ out, int P) {
  int i = blockIdx.x * 256 + threadIdx.x;
  if (i < P) out[i] = 1.f / (1.f + __expf(-0.5f * (t[i] + t[i + P])));
}

// =====================================================================
// Host orchestration
// =====================================================================
struct GcnP { const float *Wc, *Wn, *We, *q, *b; };

extern "C" void kernel_launch(void* const* d_in, const int* in_sizes, int n_in,
                              void* d_out, int out_size, void* d_ws, size_t ws_size,
                              hipStream_t stream) {
  (void)in_sizes; (void)n_in; (void)out_size; (void)ws_size;
  const int N0 = 10000, N1 = 10000, D = 512, P = 4096;
  const size_t NH = (size_t)N0 * D;

  const float* x0   = (const float*)d_in[0];
  const int*   adj0 = (const int*)  d_in[1];
  const float* e0   = (const float*)d_in[2];
  const float* x1   = (const float*)d_in[3];
  const int*   adj1 = (const int*)  d_in[4];
  const float* e1   = (const float*)d_in[5];
  const int*   ex   = (const int*)  d_in[6];
  auto gp = [&](int g) {
    GcnP p; int b = 7 + g * 5;
    p.Wc = (const float*)d_in[b]; p.Wn = (const float*)d_in[b + 1];
    p.We = (const float*)d_in[b + 2]; p.q = (const float*)d_in[b + 3];
    p.b  = (const float*)d_in[b + 4]; return p;
  };
  GcnP g10 = gp(0), g11 = gp(1), g20 = gp(2), g21 = gp(3), gf0 = gp(4), gf1 = gp(5);
  const float* A_p    = (const float*)d_in[37];
  const float* crossW = (const float*)d_in[38];
  const float* crossB = (const float*)d_in[39];
  const float* d1W    = (const float*)d_in[40];
  const float* d1B    = (const float*)d_in[41];
  const float* d2W    = (const float*)d_in[42];
  const float* d2B    = (const float*)d_in[43];
  const float* bilW   = (const float*)d_in[44];
  const float* bilB   = (const float*)d_in[45];

  float* out_head = (float*)d_out;        // 4096
  float* out_l0   = out_head + P;         // 20000
  float* out_l1   = out_l0 + 2 * N0;      // 20000
  float* aff      = out_l1 + 2 * N1;      // 1e8 (scratch in iter 1, final in iter 2)

  // workspace allocator
  char* wsb = (char*)d_ws; size_t off = 0;
  auto alloc = [&](size_t bytes) -> void* {
    void* p = wsb + off; off += (bytes + 255) & ~(size_t)255; return p;
  };
  float* zc   = (float*)alloc(NH * 4);
  float* vw   = (float*)alloc(NH * 4);
  float* vq   = (float*)alloc((size_t)N0 * 4);
  float* weq  = (float*)alloc(256);
  float* t0   = (float*)alloc(NH * 4);
  float* x00  = (float*)alloc(NH * 4);
  float* x00s = (float*)alloc(NH * 4);
  float* x10  = (float*)alloc(NH * 4);
  float* x10s = (float*)alloc(NH * 4);
  float* h0   = (float*)alloc(NH * 4);
  float* h1   = (float*)alloc(NH * 4);
  float* h0s  = (float*)alloc(NH * 4);
  float* h1s  = (float*)alloc(NH * 4);
  float* Tm   = (float*)alloc(NH * 4);
  float* enc0 = (float*)alloc(NH * 4);
  float* enc1 = (float*)alloc(NH * 4);
  float* hc   = (float*)alloc(NH * 4);
  float* dense1 = (float*)alloc((size_t)8192 * 512 * 4);
  float* tmp8   = (float*)alloc(8192 * 4);
  float* Asym   = (float*)alloc((size_t)512 * 512 * 4);
  float* c512   = (float*)alloc(512 * 4);
  float* u512   = (float*)alloc(512 * 4);
  float* rmax   = (float*)alloc((size_t)N0 * 4);
  float* rinv   = (float*)alloc((size_t)N0 * 4);
  __bf16* ab    = (__bf16*)alloc((size_t)N0 * 1024 * 2);
  __bf16* wbT   = (__bf16*)alloc((size_t)512 * 512 * 2);
  __bf16* wb2T  = (__bf16*)alloc((size_t)512 * 512 * 2);
  __bf16* h0b   = (__bf16*)alloc(NH * 2);
  __bf16* h1b   = (__bf16*)alloc(NH * 2);
  __bf16* h0bT  = (__bf16*)alloc(NH * 2);
  __bf16* h1bT  = (__bf16*)alloc(NH * 2);
  __bf16* tb    = (__bf16*)alloc(NH * 2);
  __bf16* asbT  = (__bf16*)alloc((size_t)512 * 512 * 2);
  __bf16* cwbT  = (__bf16*)alloc((size_t)1024 * 512 * 2);
  __bf16* d1wbT = (__bf16*)alloc((size_t)1024 * 512 * 2);

  auto conv = [&](const float* in, __bf16* out, size_t n) {
    k_f2bf<<<(unsigned)((n + 255) / 256), 256, 0, stream>>>(in, out, n);
  };
  auto convT = [&](const float* in, __bf16* out, int R, int C) {
    k_f2bfT<<<(unsigned)(((size_t)R * C + 255) / 256), 256, 0, stream>>>(in, out, R, C);
  };
  // C[M,N] = A(bf16,[M,K],lda) @ Bt(bf16,[N,K],ldb)^T
  auto gemm = [&](const __bf16* A, const __bf16* Bt, float* C, int M, int N, int K,
                  int lda, int ldb, const float* bias, int relu) {
    dim3 g((N + 127) / 128, (M + 127) / 128);
    gemm_wmma<0><<<g, 256, 0, stream>>>((const void*)A, Bt, C, M, N, K, lda, ldb, N,
                                        bias, relu, nullptr, nullptr);
  };
  auto gemm_sm = [&](const float* A, const __bf16* Bt, float* C, int M, int N, int K,
                     int lda, int ldb, const float* mx, const float* iv, int trans) {
    dim3 g((N + 127) / 128, (M + 127) / 128);
    if (!trans)
      gemm_wmma<1><<<g, 256, 0, stream>>>((const void*)A, Bt, C, M, N, K, lda, ldb, N,
                                          nullptr, 0, mx, iv);
    else
      gemm_wmma<2><<<g, 256, 0, stream>>>((const void*)A, Bt, C, M, N, K, lda, ldb, N,
                                          nullptr, 0, mx, iv);
  };
  auto mv = [&](const float* W, const float* v, float* o, int rows, int cols,
                const float* bias, int sig) {
    k_matvec<<<(rows + 7) / 8, 256, 0, stream>>>(W, v, o, rows, cols, bias, sig);
  };
  auto run_gcn = [&](const float* x, GcnP p, const int* adj, const float* e,
                     float* outp, int permMul, float* outs) {
    conv(x, ab, NH);
    convT(p.Wc, wbT, 512, 512);
    gemm(ab, wbT, zc, N0, D, D, D, D, nullptr, 0);
    convT(p.Wn, wb2T, 512, 512);
    gemm(ab, wb2T, vw, N0, D, D, D, D, nullptr, 0);
    mv(vw, p.q, vq, N0, D, nullptr, 0);
    mv(p.We, p.q, weq, 2, D, nullptr, 0);
    k_gcn_attn<<<N0, 256, 0, stream>>>(vw, vq, zc, adj, e, weq, p.We, p.b, outp, N0, D, permMul);
    if (outs)
      k_gcn_attn<<<N0, 256, 0, stream>>>(vw, vq, zc, adj, e, weq, p.We, p.b, outs, N0, D, 7919);
  };
  auto addv = [&](const float* a, const float* b, float* o) {
    k_add<<<(unsigned)((NH + 255) / 256), 256, 0, stream>>>(a, b, o, NH);
  };
  auto disc = [&](const float* h, const float* hs, float* outp) {
    k_colmean_relu<<<2, 256, 0, stream>>>(h, c512, N0, D);
    mv(bilW, c512, u512, D, D, nullptr, 0);          // u = W[0] @ c
    mv(h,  u512, outp,      N0, D, bilB, 1);         // sigmoid(h @ u + b)
    mv(hs, u512, outp + N0, N0, D, bilB, 1);
  };

  // ---- GCN layer 1 (GEMMs shared between plain and shuffled branches) ----
  run_gcn(x0, g10, adj0, e0, x00, 0, x00s);
  run_gcn(x1, g11, adj1, e1, x10, 0, x10s);
  // ---- GCN layer 2 + residual ----
  run_gcn(x00,  g20, adj0, e0, t0, 0,    nullptr); addv(x00,  t0, h0);
  run_gcn(x00s, g20, adj0, e0, t0, 7919, nullptr); addv(x00s, t0, h0s);
  run_gcn(x10,  g21, adj1, e1, t0, 0,    nullptr); addv(x10,  t0, h1);
  run_gcn(x10s, g21, adj1, e1, t0, 7919, nullptr); addv(x10s, t0, h1s);
  // ---- discriminators ----
  disc(h0, h0s, out_l0);
  disc(h1, h1s, out_l1);
  // ---- cross-attention loop ----
  k_asym<<<(512 * 512 + 255) / 256, 256, 0, stream>>>(A_p, Asym, 512);
  convT(Asym, asbT, 512, 512);
  convT(crossW, cwbT, 1024, 512);
  convT(d1W, d1wbT, 1024, 512);
  for (int it = 0; it < 2; ++it) {
    conv(h0, h0b, NH);                       // A operand for T = h0 @ A_sym
    conv(h1, h1b, NH);                       // [N1,K] rows for aff NT
    convT(h1, h1bT, N1, D);                  // [512,N1] B for enc0
    convT(h0, h0bT, N0, D);                  // [512,N0] B for enc1
    gemm(h0b, asbT, Tm, N0, D, D, D, D, nullptr, 0);      // T = h0 @ A_sym
    conv(Tm, tb, NH);
    gemm(tb, h1b, aff, N0, N1, D, D, D, nullptr, 0);      // aff = T @ h1^T (f32)
    k_rowstats<<<N0, 256, 0, stream>>>(aff, N1, N1, rmax, rinv);
    gemm_sm(aff, h1bT, enc0, N0, D, N1, N1, N1, rmax, rinv, 0);  // row softmax fused
    k_colstats<<<(N1 + 255) / 256, 256, 0, stream>>>(aff, N0, N1, N1, rmax, rinv);
    gemm_sm(aff, h0bT, enc1, N1, D, N0, N1, N0, rmax, rinv, 1);  // col softmax fused
    // h0 <- gcnf(relu(concat(h0,enc0) @ crossW + b))
    k_concat_bf<<<(unsigned)((NH * 2 + 255) / 256), 256, 0, stream>>>(h0, enc0, ab, N0, D);
    gemm(ab, cwbT, hc, N0, D, 1024, 1024, 1024, crossB, 1);
    run_gcn(hc, gf0, adj0, e0, h0, 0, nullptr);
    // h1 <- gcnf(relu(concat(h1,enc1) @ crossW + b))
    k_concat_bf<<<(unsigned)((NH * 2 + 255) / 256), 256, 0, stream>>>(h1, enc1, ab, N1, D);
    gemm(ab, cwbT, hc, N1, D, 1024, 1024, 1024, crossB, 1);
    run_gcn(hc, gf1, adj1, e1, h1, 0, nullptr);
  }
  // ---- head ----
  k_head_gather<<<(unsigned)(((size_t)2 * P * 1024 + 255) / 256), 256, 0, stream>>>(
      h0, h1, ex, ab, P, D);
  gemm(ab, d1wbT, dense1, 2 * P, D, 1024, 1024, 1024, d1B, 1);
  mv(dense1, d2W, tmp8, 2 * P, D, d2B, 0);
  k_head_final<<<(P + 255) / 256, 256, 0, stream>>>(tmp8, out_head, P);
}
